// BertKVMN_64819646431335
// MI455X (gfx1250) — matrix operand order repaired and checked by reading, not compiled, not run on hardware
//
#include <hip/hip_runtime.h>
#include <hip/hip_bf16.h>

// Problem constants (from reference)
#define Bn 8
#define Ln 128
#define Hn 768
#define NLAB 3

typedef __attribute__((ext_vector_type(16))) _Float16 v16h;
typedef __attribute__((ext_vector_type(8)))  _Float16 v8h;
typedef __attribute__((ext_vector_type(8)))  float    v8f;

// ---------------------------------------------------------------------------
// Kernel 1: pack valid tokens to front, zero tail, multiply by aspect mask,
//           and emit f16 directly (hidden feeds only the WMMA GEMM).
// ---------------------------------------------------------------------------
__global__ void pack_kernel(const float* __restrict__ seq,
                            const int*   __restrict__ valid,
                            const int*   __restrict__ asp,
                            _Float16*    __restrict__ hidden)   // [B,L,H] f16
{
    const int b   = blockIdx.x;
    const int tid = threadIdx.x;           // 256 threads
    __shared__ int pos[Ln];
    __shared__ int cnt;
    if (tid == 0) {
        int c = 0;
        for (int i = 0; i < Ln; ++i)
            if (valid[b * Ln + i]) pos[c++] = i;
        cnt = c;
    }
    __syncthreads();
    const int count = cnt;
    for (int e = tid; e < Ln * Hn; e += 256) {
        const int i = e / Hn;
        const int h = e - i * Hn;
        float v = 0.0f;
        if (i < count)
            v = seq[((size_t)b * Ln + pos[i]) * Hn + h];
        hidden[((size_t)b * Ln + i) * Hn + h] =
            (_Float16)(v * (float)asp[b * Ln + i]);
    }
}

// ---------------------------------------------------------------------------
// Kernel 2: gather key rows once, zero-mask padding row, convert to f16.
//           Each gathered row is reused by 8 q-tiles in the GEMM, so paying
//           the f32->f16 conversion once here halves GEMM global traffic and
//           removes the divergent padding branch from the WMMA hot loop.
// ---------------------------------------------------------------------------
__global__ void keycvt_kernel(const float* __restrict__ W_key,  // [VOCAB,H]
                              const int*   __restrict__ kv_idx, // [B,L]
                              _Float16*    __restrict__ keyemb) // [B,L,H] f16
{
    const int row = blockIdx.x;            // b*L + k
    const int tid = threadIdx.x;           // 256 threads
    const int ridx = kv_idx[row];
    const float* src = W_key + (size_t)ridx * Hn;
    _Float16* dst = keyemb + (size_t)row * Hn;
    if (ridx == 0) {
        for (int h = tid; h < Hn; h += 256) dst[h] = (_Float16)0.0f;
    } else {
        for (int h = tid; h < Hn; h += 256) dst[h] = (_Float16)src[h];
    }
}

// ---------------------------------------------------------------------------
// Kernel 3: u[b,q,k] = dot(hidden[b,q,:], keyemb[b,k,:]) / sqrt(H)
//   One wave per 16(q) x 64(k) output strip: 4 f32 accumulators, one A
//   fragment feeds 4 back-to-back v_wmma_f32_16x16x32_f16 ops per K step.
//   Fragment layouts used here (contiguous k order, same for A and B):
//     A 16x32: lane -> row M=lane&15; elem i -> K=half*16+i
//     B 32x16: lane -> col N=lane&15; elem i -> K=half*16+i
//     C 16x16: vgpr v -> row M=half*8+v, col N=lane&15
// ---------------------------------------------------------------------------
__global__ __launch_bounds__(32)
void gemm_u_kernel(const _Float16* __restrict__ hidden,  // [B,L,H] f16
                   const _Float16* __restrict__ keyemb,  // [B,L,H] f16
                   float*          __restrict__ u)       // [B,L,L]
{
    const int lane = threadIdx.x;          // wave32, EXEC all ones
    const int ktg = blockIdx.x;            // group of 4 k tiles (64 cols)
    const int qt  = blockIdx.y;            // q tile
    const int b   = blockIdx.z;
    const int half = lane >> 4;
    const int mn   = lane & 15;

    const _Float16* Ap  = hidden + ((size_t)b * Ln + qt * 16 + mn) * Hn;
    const _Float16* Bp0 = keyemb + ((size_t)b * Ln + ktg * 64 +  0 + mn) * Hn;
    const _Float16* Bp1 = keyemb + ((size_t)b * Ln + ktg * 64 + 16 + mn) * Hn;
    const _Float16* Bp2 = keyemb + ((size_t)b * Ln + ktg * 64 + 32 + mn) * Hn;
    const _Float16* Bp3 = keyemb + ((size_t)b * Ln + ktg * 64 + 48 + mn) * Hn;

    v8f c0 = {}, c1 = {}, c2 = {}, c3 = {};
    for (int kk = 0; kk < Hn; kk += 32) {
        const v16h a = *(const v16h*)(Ap + kk + half * 16);

        const v16h b0 = *(const v16h*)(Bp0 + kk + half * 16);
        const v16h b1 = *(const v16h*)(Bp1 + kk + half * 16);
        const v16h b2 = *(const v16h*)(Bp2 + kk + half * 16);
        const v16h b3 = *(const v16h*)(Bp3 + kk + half * 16);

        c0 = __builtin_amdgcn_wmma_f32_16x16x32_f16(false, a, false, b0, (short)0, c0, false, false);
        c1 = __builtin_amdgcn_wmma_f32_16x16x32_f16(false, a, false, b1, (short)0, c1, false, false);
        c2 = __builtin_amdgcn_wmma_f32_16x16x32_f16(false, a, false, b2, (short)0, c2, false, false);
        c3 = __builtin_amdgcn_wmma_f32_16x16x32_f16(false, a, false, b3, (short)0, c3, false, false);
    }

    const float scale = 0.03608439182435161f;   // 1/sqrt(768)
    float* urow = u + ((size_t)b * Ln + qt * 16 + half * 8) * Ln + ktg * 64 + mn;
    #pragma unroll
    for (int v = 0; v < 8; ++v) {
        urow[(size_t)v * Ln +  0] = c0[v] * scale;
        urow[(size_t)v * Ln + 16] = c1[v] * scale;
        urow[(size_t)v * Ln + 32] = c2[v] * scale;
        urow[(size_t)v * Ln + 48] = c3[v] * scale;
    }
}

// ---------------------------------------------------------------------------
// Kernel 4: p = exp(u)*pos / (sum_k + 1e-10), in place over u
// ---------------------------------------------------------------------------
__global__ void softmax_pos_kernel(float* __restrict__ u,       // [B*L, L]
                                   const int* __restrict__ pos) // [B*L, L]
{
    const int row = blockIdx.x;            // b*L + q
    const int k   = threadIdx.x;           // 128 threads
    const float d = __expf(u[(size_t)row * Ln + k]) *
                    (float)pos[(size_t)row * Ln + k];
    __shared__ float s[Ln];
    s[k] = d;
    __syncthreads();
    for (int off = Ln / 2; off > 0; off >>= 1) {
        if (k < off) s[k] += s[k + off];
        __syncthreads();
    }
    u[(size_t)row * Ln + k] = d / (s[0] + 1e-10f);
}

// ---------------------------------------------------------------------------
// Kernel 5: o[b,q,h] = sum_k p[b,q,k] * W_val[features[b,q,k], h]
//   L2-resident gather; uniform skip of p==0 rows (exact), prefetch next row.
// ---------------------------------------------------------------------------
__global__ void gather_o_kernel(const float* __restrict__ p,        // [B*L, L]
                                const int*   __restrict__ features, // [B*L, L]
                                const float* __restrict__ W_val,    // [FEAT,H]
                                float*       __restrict__ o)        // [B*L, H]
{
    const int row = blockIdx.x;            // b*L + q
    const int tid = threadIdx.x;           // 256 threads -> 3 h each
    __shared__ float sp[Ln];
    __shared__ int   sf[Ln];
    if (tid < Ln) {
        sp[tid] = p[(size_t)row * Ln + tid];
        sf[tid] = features[(size_t)row * Ln + tid];
    }
    __syncthreads();

    const int h0 = tid, h1 = tid + 256, h2 = tid + 512;
    float a0 = 0.f, a1 = 0.f, a2 = 0.f;
    for (int k = 0; k < Ln; ++k) {
        const float pv = sp[k];
        if (k + 1 < Ln) {   // warm L2/L0 for the next gathered row
            __builtin_prefetch(W_val + (size_t)sf[k + 1] * Hn + h0, 0, 1);
        }
        if (pv != 0.0f) {   // pos mask -> exact zero, uniform branch
            const int f = sf[k];
            if (f != 0) {   // padding_idx row is zero
                const float* vr = W_val + (size_t)f * Hn;
                a0 += pv * vr[h0];
                a1 += pv * vr[h1];
                a2 += pv * vr[h2];
            }
        }
    }
    float* orow = o + (size_t)row * Hn;
    orow[h0] = a0; orow[h1] = a1; orow[h2] = a2;
}

// ---------------------------------------------------------------------------
// Kernel 6: avg_o[b,h] = sum_q o[b,q,h] / count_q(o[b,q,h] != 0)
// ---------------------------------------------------------------------------
__global__ void avg_kernel(const float* __restrict__ o,   // [B,L,H]
                           float*       __restrict__ avg) // [B,H]
{
    const int b   = blockIdx.x;
    const int tid = threadIdx.x;           // 256 threads
    for (int h = tid; h < Hn; h += 256) {
        float s = 0.f, cnt = 0.f;
        for (int q = 0; q < Ln; ++q) {
            const float v = o[((size_t)b * Ln + q) * Hn + h];
            s += v;
            cnt += (v != 0.0f) ? 1.0f : 0.0f;
        }
        avg[b * Hn + h] = s / cnt;         // reference has no epsilon here
    }
}

// ---------------------------------------------------------------------------
// Kernel 7: logits[b,n] = dot(concat(pooled[b], avg[b]), W_dense[n]) + b[n]
// ---------------------------------------------------------------------------
__global__ void head_kernel(const float* __restrict__ pooled, // [B,H]
                            const float* __restrict__ avg,    // [B,H]
                            const float* __restrict__ Wd,     // [3,2H]
                            const float* __restrict__ bd,     // [3]
                            float*       __restrict__ out)    // [B,3]
{
    const int b = blockIdx.x / NLAB;
    const int n = blockIdx.x % NLAB;
    const int tid = threadIdx.x;           // 256 threads
    const float* w = Wd + (size_t)n * (2 * Hn);
    float s = 0.f;
    for (int i = tid; i < 2 * Hn; i += 256) {
        const float x = (i < Hn) ? pooled[b * Hn + i]
                                 : avg[b * Hn + (i - Hn)];
        s += x * w[i];
    }
    __shared__ float sh[256];
    sh[tid] = s;
    __syncthreads();
    for (int off = 128; off > 0; off >>= 1) {
        if (tid < off) sh[tid] += sh[tid + off];
        __syncthreads();
    }
    if (tid == 0) out[b * NLAB + n] = sh[0] + bd[n];
}

// ---------------------------------------------------------------------------
extern "C" void kernel_launch(void* const* d_in, const int* in_sizes, int n_in,
                              void* d_out, int out_size, void* d_ws, size_t ws_size,
                              hipStream_t stream)
{
    const float* seq     = (const float*)d_in[0];   // [B,L,H]
    const float* pooled  = (const float*)d_in[1];   // [B,H]
    const float* W_key   = (const float*)d_in[2];   // [VOCAB,H]
    const float* W_val   = (const float*)d_in[3];   // [FEAT_VOCAB,H]
    const float* W_dense = (const float*)d_in[4];   // [3,2H]
    const float* b_dense = (const float*)d_in[5];   // [3]
    const int*   valid   = (const int*)d_in[6];     // [B,L]
    const int*   kv_idx  = (const int*)d_in[7];     // [B,L]
    const int*   feats   = (const int*)d_in[8];     // [B,L,L]
    const int*   posm    = (const int*)d_in[9];     // [B,L,L]
    const int*   aspect  = (const int*)d_in[10];    // [B,L]
    float* out = (float*)d_out;                     // [B,3]

    // Workspace layout
    _Float16* hidden = (_Float16*)d_ws;                       // B*L*H f16
    _Float16* keyemb = hidden + (size_t)Bn * Ln * Hn;         // B*L*H f16
    float*    u      = (float*)(keyemb + (size_t)Bn * Ln * Hn); // B*L*L f32 (u -> p in place)
    float*    o      = u + (size_t)Bn * Ln * Ln;              // B*L*H f32
    float*    avg    = o + (size_t)Bn * Ln * Hn;              // B*H   f32

    // 1) pack + aspect -> f16
    pack_kernel<<<Bn, 256, 0, stream>>>(seq, valid, aspect, hidden);
    // 2) gather + zero-mask + f16 convert key rows
    keycvt_kernel<<<Bn * Ln, 256, 0, stream>>>(W_key, kv_idx, keyemb);
    // 3) WMMA batched GEMM: u = hidden . keyemb^T / sqrt(H), 16x64 per wave
    gemm_u_kernel<<<dim3(Ln / 64, Ln / 16, Bn), 32, 0, stream>>>(hidden, keyemb, u);
    // 4) masked exp-normalize (in place u -> p)
    softmax_pos_kernel<<<Bn * Ln, Ln, 0, stream>>>(u, posm);
    // 5) value gather-reduce
    gather_o_kernel<<<Bn * Ln, 256, 0, stream>>>(u, feats, W_val, o);
    // 6) nonzero-count average over q
    avg_kernel<<<Bn, 256, 0, stream>>>(o, avg);
    // 7) classification head
    head_kernel<<<Bn * NLAB, 256, 0, stream>>>(pooled, avg, W_dense, b_dense, out);
}
